// MultiHeadAttention_1597727834147
// MI455X (gfx1250) — matrix operand order repaired and checked by reading, not compiled, not run on hardware
//
#include <hip/hip_runtime.h>
#include <stdint.h>

// ---------- CDNA5 WMMA types ----------
typedef __attribute__((ext_vector_type(16))) __bf16 v16bf;
typedef __attribute__((ext_vector_type(8)))  __bf16 v8bf;
typedef __attribute__((ext_vector_type(8)))  float  v8f;

union AFrag { v16bf v; v8bf h[2]; };

#define WMMA_BF16(a, b, c) \
  __builtin_amdgcn_wmma_f32_16x16x32_bf16(false, (a), false, (b), (short)0, (c), false, false)

// Problem constants (from reference)
#define BATCH   2
#define TSEQ    2048
#define NEMBD   2048
#define NHEAD   16
#define NKVH    4
#define HD      128
#define MTOT    (BATCH * TSEQ)          // 4096 GEMM rows
#define QSCALE  0.08838834764831845f    // 128^-0.5
#define L2_10K_OVER_D 0.10381025296523007f // log2(10000)/128

// ---------- gfx1250 async global->LDS (ASYNCcnt path) ----------
// VFLAT GVS form: global_load_async_to_lds_b128 vdst_lds, vaddr_off32, saddr64
__device__ __forceinline__ void async_ld16(uint32_t lds_off, uint32_t voff, const void* sbase) {
  asm volatile("global_load_async_to_lds_b128 %0, %1, %2"
               :: "v"(lds_off), "v"(voff), "s"(sbase) : "memory");
}
__device__ __forceinline__ void wait_async() {
  asm volatile("s_wait_asynccnt 0x0" ::: "memory");
}

// ---------- f32 -> bf16 convert ----------
__global__ void cvt_bf16(const float* __restrict__ s, __bf16* __restrict__ d, int n) {
  int i = blockIdx.x * blockDim.x + threadIdx.x;
  int stride = gridDim.x * blockDim.x;
  for (; i < n; i += stride) d[i] = (__bf16)s[i];
}

// ---------- WMMA GEMM: C = A(M x 2048) @ W(N x 2048)^T, wave tile 32x64 ----------
// MODE 0: q proj + RoPE + scale -> q[b][h][t][d]      (bf16)
// MODE 1: k proj + RoPE         -> k[b][kvh][t][d]    (bf16)
// MODE 2: v proj                -> vT[b][kvh][d][t]   (bf16)
// MODE 3: out proj              -> out[m][n]          (f32)
template <int MODE>
__global__ __launch_bounds__(256) void gemm16(const __bf16* __restrict__ A,
                                              const __bf16* __restrict__ W,
                                              int NT64, void* __restrict__ dst) {
  const int wave = threadIdx.x >> 5;
  const int lane = threadIdx.x & 31;
  const int l15  = lane & 15;
  const int lh   = lane >> 4;
  const int wid  = blockIdx.x * 8 + wave;
  const int m0   = (wid / NT64) * 32;
  const int n0   = (wid % NT64) * 64;

  const __bf16* __restrict__ arow0 = A + (size_t)(m0 + l15) * NEMBD;
  const __bf16* __restrict__ arow1 = arow0 + (size_t)16 * NEMBD;
  const __bf16* __restrict__ brow  = W + (size_t)(n0 + l15) * NEMBD;

  v8f acc[2][4];
  const v8f zero = {};
#pragma unroll
  for (int mi = 0; mi < 2; ++mi)
#pragma unroll
    for (int j = 0; j < 4; ++j) acc[mi][j] = zero;

  for (int kb = 0; kb < NEMBD; kb += 32) {
    __builtin_prefetch(arow0 + kb + 256, 0, 1);
    __builtin_prefetch(arow1 + kb + 256, 0, 1);
    __builtin_prefetch(brow + kb + 256, 0, 1);
    AFrag a0, a1;  // A layout: row = lane%16, k = (lane/16)*16 + g*8 + o
    a0.h[0] = *(const v8bf*)(arow0 + kb + lh * 16);
    a0.h[1] = *(const v8bf*)(arow0 + kb + 8 + lh * 16);
    a1.h[0] = *(const v8bf*)(arow1 + kb + lh * 16);
    a1.h[1] = *(const v8bf*)(arow1 + kb + 8 + lh * 16);
#pragma unroll
    for (int j = 0; j < 4; ++j) {
      // B layout: col = lane%16, k = (lane/16)*16 + i  (row of W = col of W^T)
      v16bf bw = *(const v16bf*)(brow + (size_t)j * 16 * NEMBD + kb + lh * 16);
      acc[0][j] = WMMA_BF16(a0.v, bw, acc[0][j]);
      acc[1][j] = WMMA_BF16(a1.v, bw, acc[1][j]);
    }
  }

  // Epilogue.  C layout: row m = r + 8*(lane/16), col n = lane%16.
#pragma unroll
  for (int mi = 0; mi < 2; ++mi) {
#pragma unroll
    for (int j = 0; j < 4; ++j) {
      const int n_g = n0 + j * 16 + l15;
#pragma unroll
      for (int r = 0; r < 8; ++r) {
        float val = acc[mi][j][r];
        const int m_g = m0 + mi * 16 + r + 8 * lh;
        if (MODE == 0 || MODE == 1) {
          // RoPE: pair element (d^1) lives in the adjacent lane of this C tile.
          float partner = __shfl_xor(val, 1, 32);
          const int d = n_g & (HD - 1);
          const int head = n_g >> 7;
          const int t = m_g & (TSEQ - 1);
          const int bb = m_g >> 11;
          const int p2 = d & ~1;
          const float ang = (float)t * exp2f(-(float)p2 * L2_10K_OVER_D);
          const float cs = cosf(ang), sn = sinf(ang);
          float res = val * cs + partner * ((d & 1) ? sn : -sn);
          if (MODE == 0) {
            res *= QSCALE;
            ((__bf16*)dst)[(((size_t)(bb * NHEAD + head)) * TSEQ + t) * HD + d] = (__bf16)res;
          } else {
            ((__bf16*)dst)[(((size_t)(bb * NKVH + head)) * TSEQ + t) * HD + d] = (__bf16)res;
          }
        } else if (MODE == 2) {
          const int d = n_g & (HD - 1);
          const int head = n_g >> 7;
          const int t = m_g & (TSEQ - 1);
          const int bb = m_g >> 11;
          ((__bf16*)dst)[(((size_t)(bb * NKVH + head)) * HD + d) * TSEQ + t] = (__bf16)val;
        } else {
          ((float*)dst)[(size_t)m_g * NEMBD + n_g] = val;
        }
      }
    }
  }
}

// ---------- Flash attention ----------
// Block = 8 waves = 8 consecutive 16-query tiles of the same (b, head).
// K/V blocks (32 keys) staged into LDS with async global->LDS DMA, double
// buffered so the copy of block kb+1 overlaps the WMMA compute of block kb.
__global__ __launch_bounds__(256) void attn_kernel(const __bf16* __restrict__ qg,
                                                   const __bf16* __restrict__ kg,
                                                   const __bf16* __restrict__ vtg,
                                                   const unsigned char* __restrict__ amask,
                                                   __bf16* __restrict__ og) {
  __shared__ __attribute__((aligned(64))) __bf16 kbuf[2][32 * HD];   // [key][d], 2 x 8KB
  __shared__ __attribute__((aligned(64))) __bf16 vbuf[2][HD * 32];   // [d][key], 2 x 8KB
  __shared__ __attribute__((aligned(32))) __bf16 pbuf[8][16 * 32];   // per-wave P transpose

  const int wave = threadIdx.x >> 5;
  const int lane = threadIdx.x & 31;
  const int l15  = lane & 15;
  const int lh   = lane >> 4;
  const int grp  = blockIdx.x;          // 512 groups
  const int qt0  = (grp & 15) * 8;      // first q-tile of this block
  const int h    = (grp >> 4) & 15;
  const int b    = grp >> 8;
  const int qt   = qt0 + wave;
  const int qm0  = qt * 16;
  const int hkv  = h >> 2;              // groups = 4

  // Preload Q tile (16 x 128) as 4 A-fragments.
  const __bf16* __restrict__ qrow =
      qg + ((size_t)(b * NHEAD + h) * TSEQ + (qm0 + l15)) * HD;
  AFrag qa[4];
#pragma unroll
  for (int ks = 0; ks < 4; ++ks) {
    qa[ks].h[0] = *(const v8bf*)(qrow + ks * 32 + lh * 16);
    qa[ks].h[1] = *(const v8bf*)(qrow + ks * 32 + 8 + lh * 16);
  }
  const __bf16* __restrict__ kbase = kg  + (size_t)(b * NKVH + hkv) * TSEQ * HD;
  const __bf16* __restrict__ vbase = vtg + (size_t)(b * NKVH + hkv) * HD * TSEQ;

  const uint32_t klds0 = (uint32_t)(uintptr_t)&kbuf[0][0];
  const uint32_t vlds0 = (uint32_t)(uintptr_t)&vbuf[0][0];
  const int c0 = wave * 2, c1 = c0 + 1;            // 2 of 16 chunks per wave
  const uint32_t laneOff = (uint32_t)lane * 16;    // K: 512B contiguous chunks
  const uint32_t vr = (uint32_t)(lane >> 2);       // V: 8 rows x 64B per chunk
  const uint32_t vc = (uint32_t)(lane & 3) * 16;

  v8f o[8];
  const v8f zero = {};
#pragma unroll
  for (int dt = 0; dt < 8; ++dt) o[dt] = zero;
  float mrow[8], lrow[8];
#pragma unroll
  for (int r = 0; r < 8; ++r) { mrow[r] = -1e30f; lrow[r] = 0.0f; }

  // All waves run to the causal limit of the LAST tile in the group; the
  // element-wise mask makes the extra blocks exact no-ops for earlier tiles.
  const int nkb = ((qt0 * 16 + 127) >> 5) + 1;

  // Cooperative async stage of one 32-key K/V block into buffer `buf`.
  auto stage = [&](int key0, int buf) {
    const uint32_t klds = klds0 + (uint32_t)buf * (32 * HD * 2);
    const uint32_t vlds = vlds0 + (uint32_t)buf * (HD * 32 * 2);
    const uint32_t kg0 = (uint32_t)key0 * 256;     // K rows contiguous: 8KB block
    async_ld16(klds + c0 * 512 + laneOff, kg0 + c0 * 512 + laneOff, kbase);
    async_ld16(klds + c1 * 512 + laneOff, kg0 + c1 * 512 + laneOff, kbase);
    const uint32_t vg0 = (uint32_t)key0 * 2;       // V^T rows: 64B slices, stride 4096B
    const uint32_t d0 = (uint32_t)c0 * 8 + vr, d1 = (uint32_t)c1 * 8 + vr;
    async_ld16(vlds + d0 * 64 + vc, d0 * 4096 + vg0 + vc, vbase);
    async_ld16(vlds + d1 * 64 + vc, d1 * 4096 + vg0 + vc, vbase);
  };

  stage(0, 0);
  for (int kb = 0; kb < nkb; ++kb) {
    const int cur = kb & 1;
    const int key0 = kb * 32;
    wait_async();        // my chunks of block kb landed
    __syncthreads();     // everyone's chunks landed; everyone done with buf cur^1
    if (kb + 1 < nkb) stage(key0 + 32, cur ^ 1);   // overlap next DMA with compute

    // S = Q @ K^T  (two 16x16 column tiles, K-dim = 128 = 4 x 32), K from LDS.
    v8f s0 = zero, s1 = zero;
#pragma unroll
    for (int ks = 0; ks < 4; ++ks) {
      v16bf b0 = *(const v16bf*)&kbuf[cur][(size_t)l15 * HD + ks * 32 + lh * 16];
      v16bf b1 = *(const v16bf*)&kbuf[cur][(size_t)(16 + l15) * HD + ks * 32 + lh * 16];
      s0 = WMMA_BF16(qa[ks].v, b0, s0);
      s1 = WMMA_BF16(qa[ks].v, b1, s1);
    }
    const int keyA = key0 + l15;
    const int keyB = key0 + 16 + l15;
    const bool okA = amask[b * TSEQ + keyA] != 0;
    const bool okB = amask[b * TSEQ + keyB] != 0;

    float alpha[8];
#pragma unroll
    for (int r = 0; r < 8; ++r) {
      const int qr = qm0 + r + 8 * lh;
      float v0 = (keyA <= qr && okA) ? s0[r] : -1e30f;
      float v1 = (keyB <= qr && okB) ? s1[r] : -1e30f;
      float rm = fmaxf(v0, v1);
      rm = fmaxf(rm, __shfl_xor(rm, 1, 32));
      rm = fmaxf(rm, __shfl_xor(rm, 2, 32));
      rm = fmaxf(rm, __shfl_xor(rm, 4, 32));
      rm = fmaxf(rm, __shfl_xor(rm, 8, 32));
      const float mnew = fmaxf(mrow[r], rm);
      const float al = __expf(mrow[r] - mnew);
      const float p0 = __expf(v0 - mnew);
      const float p1 = __expf(v1 - mnew);
      float rs = p0 + p1;
      rs += __shfl_xor(rs, 1, 32);
      rs += __shfl_xor(rs, 2, 32);
      rs += __shfl_xor(rs, 4, 32);
      rs += __shfl_xor(rs, 8, 32);
      lrow[r] = lrow[r] * al + rs;
      mrow[r] = mnew;
      alpha[r] = al;
      // Park P (C layout) in this wave's LDS strip for the layout change.
      const int prow = r + 8 * lh;
      pbuf[wave][prow * 32 + l15]      = (__bf16)p0;
      pbuf[wave][prow * 32 + 16 + l15] = (__bf16)p1;
    }
#pragma unroll
    for (int dt = 0; dt < 8; ++dt)
#pragma unroll
      for (int r = 0; r < 8; ++r) o[dt][r] *= alpha[r];

    // Re-read P in A layout (same wave: DS ops are in-order).
    AFrag pa;
    pa.h[0] = *(const v8bf*)(&pbuf[wave][l15 * 32 + lh * 16]);
    pa.h[1] = *(const v8bf*)(&pbuf[wave][l15 * 32 + 8 + lh * 16]);

    // O += P(16x32) @ V(32x128); V^T tile from LDS -> contiguous B fragments.
#pragma unroll
    for (int dt = 0; dt < 8; ++dt) {
      v16bf bv = *(const v16bf*)&vbuf[cur][(size_t)(dt * 16 + l15) * 32 + lh * 16];
      o[dt] = WMMA_BF16(pa.v, bv, o[dt]);
    }
  }

  // Normalize and emit [b*T + t][h*128 + d] (bf16 for the final GEMM).
#pragma unroll
  for (int r = 0; r < 8; ++r) {
    const float inv = 1.0f / lrow[r];
    const int t = qm0 + r + 8 * lh;
    const size_t row = (size_t)(b * TSEQ + t) * NEMBD;
#pragma unroll
    for (int dt = 0; dt < 8; ++dt)
      og[row + h * HD + dt * 16 + l15] = (__bf16)(o[dt][r] * inv);
  }
}

// ---------- launcher ----------
extern "C" void kernel_launch(void* const* d_in, const int* in_sizes, int n_in,
                              void* d_out, int out_size, void* d_ws, size_t ws_size,
                              hipStream_t stream) {
  const float* x  = (const float*)d_in[0];
  const unsigned char* amask = (const unsigned char*)d_in[1];
  const float* Wq = (const float*)d_in[2];
  const float* Wk = (const float*)d_in[3];
  const float* Wv = (const float*)d_in[4];
  const float* Wo = (const float*)d_in[5];
  float* out = (float*)d_out;

  char* p = (char*)d_ws;
  __bf16* xb  = (__bf16*)p; p += (size_t)MTOT * NEMBD * 2;             // 16.8 MB
  __bf16* wqb = (__bf16*)p; p += (size_t)NEMBD * NEMBD * 2;            //  8.4 MB
  __bf16* wkb = (__bf16*)p; p += (size_t)(NKVH * HD) * NEMBD * 2;      //  2.1 MB
  __bf16* wvb = (__bf16*)p; p += (size_t)(NKVH * HD) * NEMBD * 2;      //  2.1 MB
  __bf16* wob = (__bf16*)p; p += (size_t)NEMBD * NEMBD * 2;            //  8.4 MB
  __bf16* qb  = (__bf16*)p; p += (size_t)BATCH * NHEAD * TSEQ * HD * 2;// 16.8 MB
  __bf16* kbb = (__bf16*)p; p += (size_t)BATCH * NKVH * TSEQ * HD * 2; //  4.2 MB
  __bf16* vtb = (__bf16*)p; p += (size_t)BATCH * NKVH * HD * TSEQ * 2; //  4.2 MB
  __bf16* ob  = (__bf16*)p; p += (size_t)MTOT * NEMBD * 2;             // 16.8 MB

  cvt_bf16<<<2048, 256, 0, stream>>>(x,  xb,  MTOT * NEMBD);
  cvt_bf16<<<2048, 256, 0, stream>>>(Wq, wqb, NEMBD * NEMBD);
  cvt_bf16<<<1024, 256, 0, stream>>>(Wk, wkb, NKVH * HD * NEMBD);
  cvt_bf16<<<1024, 256, 0, stream>>>(Wv, wvb, NKVH * HD * NEMBD);
  cvt_bf16<<<2048, 256, 0, stream>>>(Wo, wob, NEMBD * NEMBD);

  // Q/K/V projections: wave tiles = (M/32)*(N/64); 8 waves/block.
  gemm16<0><<<512, 256, 0, stream>>>(xb, wqb, NEMBD / 64, (void*)qb);     // 128*32 tiles
  gemm16<1><<<128, 256, 0, stream>>>(xb, wkb, (NKVH * HD) / 64, (void*)kbb);
  gemm16<2><<<128, 256, 0, stream>>>(xb, wvb, (NKVH * HD) / 64, (void*)vtb);

  // Attention: B*H*(T/128) = 512 blocks, 8 query tiles each.
  attn_kernel<<<512, 256, 0, stream>>>(qb, kbb, vtb, amask, ob);

  // Output projection -> f32 d_out.
  gemm16<3><<<512, 256, 0, stream>>>(ob, wob, NEMBD / 64, (void*)out);
}
